// SelfAttention_56702158242293
// MI455X (gfx1250) — matrix operand order repaired and checked by reading, not compiled, not run on hardware
//
#include <hip/hip_runtime.h>
#include <hip/hip_bf16.h>

// ---------------------------------------------------------------------------
// Types for CDNA5 WMMA (wave32): v16bf operands, v8f accumulator.
// ---------------------------------------------------------------------------
typedef __attribute__((ext_vector_type(16))) __bf16 v16bf;
typedef __attribute__((ext_vector_type(8)))  __bf16 v8bf;
typedef __attribute__((ext_vector_type(4)))  __bf16 bf16x4;
typedef __attribute__((ext_vector_type(8)))  float  v8f;
typedef __attribute__((__vector_size__(4 * sizeof(int)))) int v4i;

typedef __attribute__((address_space(1))) v4i* gv4i_ptr;   // global
typedef __attribute__((address_space(3))) v4i* lv4i_ptr;   // LDS

__device__ __forceinline__ v16bf combine16(v8bf lo, v8bf hi) {
    return __builtin_shufflevector(lo, hi, 0,1,2,3,4,5,6,7,8,9,10,11,12,13,14,15);
}

__device__ __forceinline__ v8f wmma_bf16(v16bf a, v16bf b, v8f c) {
    // D = A(16x32 bf16) * B(32x16 bf16) + C(16x16 f32)
    return __builtin_amdgcn_wmma_f32_16x16x32_bf16(
        /*neg_a=*/false, a, /*neg_b=*/false, b,
        /*c_mod=*/(short)0, c, /*reuse_a=*/false, /*reuse_b=*/false);
}

__device__ __forceinline__ void wait_asynccnt0() {
#if __has_builtin(__builtin_amdgcn_s_wait_asynccnt)
    __builtin_amdgcn_s_wait_asynccnt(0);
#else
    asm volatile("s_wait_asynccnt 0x0" ::: "memory");
#endif
}

#if __has_builtin(__builtin_amdgcn_global_load_async_to_lds_b128)
__device__ __forceinline__ void async_copy_b128(const void* gsrc, void* ldst) {
    __builtin_amdgcn_global_load_async_to_lds_b128(
        (gv4i_ptr)(void*)gsrc, (lv4i_ptr)(void*)ldst, 0, 0);
}
#else
__device__ __forceinline__ void async_copy_b128(const void* gsrc, void* ldst) {
    *(v8bf*)ldst = *(const v8bf*)gsrc;   // synchronous fallback
}
#endif

// 64B (4 x b128) contiguous async copy
__device__ __forceinline__ void async_copy_b512(const __bf16* gsrc, __bf16* ldst) {
    async_copy_b128(gsrc,      ldst);
    async_copy_b128(gsrc + 8,  ldst + 8);
    async_copy_b128(gsrc + 16, ldst + 16);
    async_copy_b128(gsrc + 24, ldst + 24);
}

// ---------------------------------------------------------------------------
// Problem constants
// ---------------------------------------------------------------------------
#define BATCH     2
#define SEQ       2048
#define EMBED     2048
#define NHEADS    16
#define HEADD     128
#define QKV_COLS  (3 * EMBED)     // 6144
#define MROWS     (BATCH * SEQ)   // 4096

// ---------------------------------------------------------------------------
// One-shot precision/layout conversion kernels (amortized over GEMM reuse).
// ---------------------------------------------------------------------------
__global__ __launch_bounds__(256)
void convert_bf16(const float* __restrict__ src, __bf16* __restrict__ dst) {
    const size_t i = ((size_t)blockIdx.x * 256 + threadIdx.x) * 4;
    const float4 v = *(const float4*)(src + i);
    bf16x4 b;
    b[0] = (__bf16)v.x; b[1] = (__bf16)v.y; b[2] = (__bf16)v.z; b[3] = (__bf16)v.w;
    *(bf16x4*)(dst + i) = b;
}

// src [K][N] f32 -> dst [N][K] bf16 (32x32 tiles through LDS)
__global__ __launch_bounds__(256)
void convert_transpose_bf16(const float* __restrict__ src, __bf16* __restrict__ dst,
                            int K, int N) {
    __shared__ float tile[32][33];
    const int nb = blockIdx.x * 32;   // N base
    const int kb = blockIdx.y * 32;   // K base
    const int tx = threadIdx.x;       // 0..31
    const int ty = threadIdx.y;       // 0..7
#pragma unroll
    for (int i = 0; i < 32; i += 8)
        tile[ty + i][tx] = src[(size_t)(kb + ty + i) * N + nb + tx];
    __syncthreads();
#pragma unroll
    for (int i = 0; i < 32; i += 8)
        dst[(size_t)(nb + ty + i) * K + kb + tx] = (__bf16)tile[tx][ty + i];
}

// ---------------------------------------------------------------------------
// GEMM: C[M,N] = A[M,K](bf16) * Bt[N,K](bf16, pre-transposed) + bias[N].
// Block tile 128x128, K-step 64 (2 WMMA k-chunks), double-buffered LDS
// filled with async global->LDS b128 copies; 8 waves (4x2) -> 16 WMMAs/step.
// ---------------------------------------------------------------------------
#define GEMM_BM 128
#define GEMM_BN 128
#define GEMM_BK 64
#define AS_LD   72   // 64 + 8 pad (16B-aligned rows, spread banks)

template <bool OUT_BF16>
__global__ __launch_bounds__(256)
void gemm_bf16_wmma(const __bf16* __restrict__ A,   // [M][K] bf16
                    const __bf16* __restrict__ Bt,  // [N][K] bf16
                    const float* __restrict__ bias, // [N]
                    void* __restrict__ Cout,        // [M,N] bf16 or f32
                    int M, int N, int K) {
    __shared__ __align__(16) __bf16 As[2][GEMM_BM * AS_LD];
    __shared__ __align__(16) __bf16 Bs[2][GEMM_BN * AS_LD];

    const int tid  = threadIdx.x;
    const int lane = tid & 31;
    const int wave = tid >> 5;
    const int l16  = lane & 15;
    const int half = lane >> 4;          // 0: lanes 0-15, 1: lanes 16-31
    const int waveRow = wave & 3;        // 4 wave rows * 32 rows
    const int waveCol = wave >> 2;       // 2 wave cols * 64 cols

    const int rowBase = blockIdx.y * GEMM_BM;
    const int colBase = blockIdx.x * GEMM_BN;

    // staging assignment: thread -> (row, 32-elem segment); 8 async ops each
    const int srow = tid >> 1;           // 0..127
    const int sseg = (tid & 1) * 32;     // 0 or 32

    const __bf16* aRow = A  + (size_t)(rowBase + srow) * K + sseg;
    const __bf16* bRow = Bt + (size_t)(colBase + srow) * K + sseg;

    auto stage = [&](int kb, int buf) {
        async_copy_b512(aRow + kb * GEMM_BK, &As[buf][srow * AS_LD + sseg]);
        async_copy_b512(bRow + kb * GEMM_BK, &Bs[buf][srow * AS_LD + sseg]);
    };

    v8f acc[2][4] = {};

    const int nkb = K / GEMM_BK;
    stage(0, 0);
    for (int kb = 0; kb < nkb; ++kb) {
        const int cur = kb & 1;
        wait_asynccnt0();        // tile(kb) landed (only stage(kb) outstanding)
        __syncthreads();         // visible to all waves; prev reads finished
        if (kb + 1 < nkb) stage(kb + 1, cur ^ 1);   // overlap DMA with WMMAs

#pragma unroll
        for (int kc = 0; kc < 2; ++kc) {
            v16bf af[2];
#pragma unroll
            for (int rt = 0; rt < 2; ++rt) {
                const int arow = waveRow * 32 + rt * 16 + l16;
                const __bf16* ap = &As[cur][arow * AS_LD + kc * 32 + half * 8];
                af[rt] = combine16(*(const v8bf*)ap, *(const v8bf*)(ap + 16));
            }
#pragma unroll
            for (int cn = 0; cn < 4; ++cn) {
                const int bcol = waveCol * 64 + cn * 16 + l16;
                const __bf16* bp = &Bs[cur][bcol * AS_LD + kc * 32 + half * 8];
                const v16bf bf = combine16(*(const v8bf*)bp, *(const v8bf*)(bp + 16));
#pragma unroll
                for (int rt = 0; rt < 2; ++rt)
                    acc[rt][cn] = wmma_bf16(af[rt], bf, acc[rt][cn]);
            }
        }
    }

    // ---- epilogue: bias + store ----
#pragma unroll
    for (int cn = 0; cn < 4; ++cn) {
        const int col = colBase + waveCol * 64 + cn * 16 + l16;
        const float bv = bias[col];
#pragma unroll
        for (int rt = 0; rt < 2; ++rt) {
#pragma unroll
            for (int r = 0; r < 8; ++r) {
                const int row = rowBase + waveRow * 32 + rt * 16 + r + half * 8;
                const float v = acc[rt][cn][r] + bv;
                if (OUT_BF16)
                    ((__bf16*)Cout)[(size_t)row * N + col] = (__bf16)v;
                else
                    ((float*)Cout)[(size_t)row * N + col] = v;
            }
        }
    }
}

// ---------------------------------------------------------------------------
// Flash attention (causal). One block per (b, h, 128-query tile); 8 waves,
// each wave owns 16 query rows with Q fragments in registers. Per 64-key
// step: S = Q K^T (16 WMMAs), online softmax (one shuffle tree per 64 keys),
// O += P V (16 WMMAs). K staged with async global->LDS copies.
// ---------------------------------------------------------------------------
#define KB    64    // keys per step
#define KT_LD 136   // 128 + 8 pad
#define VT_LD 72    // 64 + 8 pad
#define PT_LD 72    // 64 + 8 pad

__global__ __launch_bounds__(256)
void flash_attn_causal(const __bf16* __restrict__ qkv,     // [B, L, 3E]
                       __bf16* __restrict__ attn_out) {    // [B, L, E] bf16
    __shared__ __align__(16) __bf16 Kt[KB * KT_LD];        // [key][d]
    __shared__ __align__(16) __bf16 Vt[HEADD * VT_LD];     // [d][key] (transposed)
    __shared__ __align__(16) __bf16 Pst[8 * 16 * PT_LD];   // per-wave P patch

    const int tid  = threadIdx.x;
    const int lane = tid & 31;
    const int wave = tid >> 5;
    const int l16  = lane & 15;
    const int half = lane >> 4;

    const int b    = blockIdx.z;
    const int h    = blockIdx.y;
    const int qblk = blockIdx.x;
    const int qbase = qblk * 128 + wave * 16;

    const float scale = 0.088388347648318447f;   // 1/sqrt(128)

    // ---- Q fragments (resident): 4 chunks of 16x32 over D=128 ----
    const __bf16* qrow = qkv + ((size_t)b * SEQ + (qbase + l16)) * QKV_COLS
                             + h * HEADD;
    v16bf qf[4];
#pragma unroll
    for (int dc = 0; dc < 4; ++dc) {
        const __bf16* p = qrow + dc * 32 + half * 8;
        qf[dc] = combine16(*(const v8bf*)p, *(const v8bf*)(p + 16));
    }

    v8f o[8] = {};
    float m[8], lsum[8];
#pragma unroll
    for (int r = 0; r < 8; ++r) { m[r] = -1e30f; lsum[r] = 0.0f; }

    const int nkb = (qblk + 1) * 2;   // 64-key blocks up to causal boundary
    for (int kb = 0; kb < nkb; ++kb) {
        __syncthreads();
        // ---- stage K tile [64 keys][128 d] via async copy ----
        {
            const int key = tid >> 2;            // 0..63
            const int ch  = (tid & 3) * 32;      // 0,32,64,96
            const __bf16* src = qkv + ((size_t)b * SEQ + kb * KB + key) * QKV_COLS
                                    + EMBED + h * HEADD + ch;
            async_copy_b512(src, &Kt[key * KT_LD + ch]);
        }
        // ---- stage V tile transposed: Vt[d][key], 64 keys ----
        {
            const int key = tid & 63;            // 0..63
            const int d0  = (tid >> 6) * 32;     // 0,32,64,96
            const __bf16* src = qkv + ((size_t)b * SEQ + kb * KB + key) * QKV_COLS
                                    + 2 * EMBED + h * HEADD + d0;
#pragma unroll
            for (int c = 0; c < 4; ++c) {
                v8bf v = *(const v8bf*)(src + c * 8);
#pragma unroll
                for (int i = 0; i < 8; ++i)
                    Vt[(d0 + c * 8 + i) * VT_LD + key] = v[i];
            }
        }
        wait_asynccnt0();
        __syncthreads();

        // ---- S = Q K^T : four 16x16 column tiles, K-dim = 128 (4 chunks) ----
        v8f s[4] = {};
#pragma unroll
        for (int dc = 0; dc < 4; ++dc) {
#pragma unroll
            for (int ct = 0; ct < 4; ++ct) {
                const __bf16* kp = &Kt[(ct * 16 + l16) * KT_LD + dc * 32 + half * 8];
                const v16bf kf = combine16(*(const v8bf*)kp, *(const v8bf*)(kp + 16));
                s[ct] = wmma_bf16(qf[dc], kf, s[ct]);
            }
        }

        // ---- online softmax over 64 keys (one shuffle tree per row) ----
        float alpha[8];
#pragma unroll
        for (int r = 0; r < 8; ++r) {
            const int q = qbase + r + half * 8;
            float a[4];
#pragma unroll
            for (int ct = 0; ct < 4; ++ct) {
                const int key = kb * KB + ct * 16 + l16;
                a[ct] = (key <= q) ? s[ct][r] * scale : -1e30f;
            }
            float mx = fmaxf(fmaxf(a[0], a[1]), fmaxf(a[2], a[3]));
#pragma unroll
            for (int off = 1; off < 16; off <<= 1)
                mx = fmaxf(mx, __shfl_xor(mx, off, 32));
            const float mn = fmaxf(m[r], mx);
            const float al = __expf(m[r] - mn);
            m[r] = mn;
            float p[4];
#pragma unroll
            for (int ct = 0; ct < 4; ++ct) p[ct] = __expf(a[ct] - mn);
            float rs = (p[0] + p[1]) + (p[2] + p[3]);
#pragma unroll
            for (int off = 1; off < 16; off <<= 1)
                rs += __shfl_xor(rs, off, 32);
            lsum[r] = lsum[r] * al + rs;
            alpha[r] = al;
            // C-layout -> memory: row = r + 8*half, col = ct*16 + l16
            __bf16* prow = &Pst[(wave * 16 + r + half * 8) * PT_LD];
#pragma unroll
            for (int ct = 0; ct < 4; ++ct)
                prow[ct * 16 + l16] = (__bf16)p[ct];
        }
#pragma unroll
        for (int t = 0; t < 8; ++t)
#pragma unroll
            for (int r = 0; r < 8; ++r)
                o[t][r] *= alpha[r];

        // ---- P fragments (A layout, 16x64 = 2 k-chunks) ----
        v16bf pf[2];
#pragma unroll
        for (int kc = 0; kc < 2; ++kc) {
            const __bf16* pp = &Pst[(wave * 16 + l16) * PT_LD + kc * 32 + half * 8];
            pf[kc] = combine16(*(const v8bf*)pp, *(const v8bf*)(pp + 16));
        }

        // ---- O += P V : 8 d-tiles x 2 k-chunks ----
#pragma unroll
        for (int t = 0; t < 8; ++t) {
#pragma unroll
            for (int kc = 0; kc < 2; ++kc) {
                const __bf16* vp = &Vt[(t * 16 + l16) * VT_LD + kc * 32 + half * 8];
                const v16bf vf = combine16(*(const v8bf*)vp, *(const v8bf*)(vp + 16));
                o[t] = wmma_bf16(pf[kc], vf, o[t]);
            }
        }
    }

    // ---- epilogue: normalize and store bf16 ----
#pragma unroll
    for (int t = 0; t < 8; ++t) {
#pragma unroll
        for (int r = 0; r < 8; ++r) {
            const int q = qbase + r + half * 8;
            const int d = t * 16 + l16;
            attn_out[((size_t)b * SEQ + q) * EMBED + h * HEADD + d] =
                (__bf16)(o[t][r] / lsum[r]);
        }
    }
}

// ---------------------------------------------------------------------------
// Launch pipeline.
// ---------------------------------------------------------------------------
extern "C" void kernel_launch(void* const* d_in, const int* in_sizes, int n_in,
                              void* d_out, int out_size, void* d_ws, size_t ws_size,
                              hipStream_t stream) {
    (void)in_sizes; (void)n_in; (void)out_size; (void)ws_size;

    const float* x     = (const float*)d_in[0];
    const float* W_in  = (const float*)d_in[1];
    const float* b_in  = (const float*)d_in[2];
    const float* W_out = (const float*)d_in[3];
    const float* b_out = (const float*)d_in[4];
    float*       out   = (float*)d_out;

    // workspace layout (bytes)
    char* ws = (char*)d_ws;
    __bf16* qkv    = (__bf16*)ws;                              // 4096*6144*2
    ws += (size_t)MROWS * QKV_COLS * 2;
    __bf16* attnbf = (__bf16*)ws;                              // 4096*2048*2
    ws += (size_t)MROWS * EMBED * 2;
    __bf16* xbf    = (__bf16*)ws;                              // 4096*2048*2
    ws += (size_t)MROWS * EMBED * 2;
    __bf16* WiT    = (__bf16*)ws;                              // 6144*2048*2
    ws += (size_t)QKV_COLS * EMBED * 2;
    __bf16* WoT    = (__bf16*)ws;                              // 2048*2048*2

    dim3 blk(256);

    // 0) one-shot conversions: x -> bf16; W_in, W_out -> transposed bf16 [N][K]
    convert_bf16<<<(unsigned)((size_t)MROWS * EMBED / 1024), blk, 0, stream>>>(x, xbf);
    convert_transpose_bf16<<<dim3(QKV_COLS / 32, EMBED / 32), dim3(32, 8), 0, stream>>>(
        W_in, WiT, EMBED, QKV_COLS);
    convert_transpose_bf16<<<dim3(EMBED / 32, EMBED / 32), dim3(32, 8), 0, stream>>>(
        W_out, WoT, EMBED, EMBED);

    // 1) QKV projection: [4096,2048] x [2048,6144] + bias -> bf16
    gemm_bf16_wmma<true><<<dim3(QKV_COLS / GEMM_BN, MROWS / GEMM_BM), blk, 0, stream>>>(
        xbf, WiT, b_in, (void*)qkv, MROWS, QKV_COLS, EMBED);

    // 2) Causal flash attention per (query-tile, head, batch)
    flash_attn_causal<<<dim3(SEQ / 128, NHEADS, BATCH), blk, 0, stream>>>(qkv, attnbf);

    // 3) Output projection: [4096,2048] x [2048,2048] + bias -> f32
    gemm_bf16_wmma<false><<<dim3(EMBED / GEMM_BN, MROWS / GEMM_BM), blk, 0, stream>>>(
        attnbf, WoT, b_out, (void*)out, MROWS, EMBED, EMBED);
}
